// PeerLookup_55473797595870
// MI455X (gfx1250) — matrix operand order repaired and checked by reading, not compiled, not run on hardware
//
#include <hip/hip_runtime.h>
#include <math.h>

// ---------------------------------------------------------------------------
// Problem constants (from reference)
// ---------------------------------------------------------------------------
#define BS      2
#define SEQ     2048
#define ROWS    (BS * SEQ)        // 4096
#define IN_F    1024
#define OUT_F   1024
#define NHEAD   4
#define QDIM    512
#define NQU     256
#define TOPK    8
#define HROWS   (ROWS * NHEAD)    // 16384

typedef unsigned short u16;
typedef __attribute__((ext_vector_type(16))) __bf16 v16bf;
typedef __attribute__((ext_vector_type(8)))  float  v8f;
typedef __attribute__((ext_vector_type(4)))  float  v4f;
typedef __attribute__((ext_vector_type(4)))  unsigned int v4u;
typedef __attribute__((ext_vector_type(4)))  int    v4i;
typedef __attribute__((ext_vector_type(4)))  u16    v4s;

#define GAS __attribute__((address_space(1)))   // global
#define LAS __attribute__((address_space(3)))   // LDS

// gfx1250 async global->LDS copy path (ASYNCcnt), with sync fallback.
// Probe round 2: builtin exists; param0 is (global int4*), param1 LDS ptr.
#if __has_builtin(__builtin_amdgcn_global_load_async_to_lds_b128) && \
    __has_builtin(__builtin_amdgcn_s_wait_asynccnt)
#define ASYNC_COPY 1
#else
#define ASYNC_COPY 0
#endif

__device__ __forceinline__ u16 f2bf(float f) {
    unsigned int u = __float_as_uint(f);
    u += 0x7fffu + ((u >> 16) & 1u);      // round-to-nearest-even
    return (u16)(u >> 16);
}

__device__ __forceinline__ float gelu_tanh(float x) {
    float x3 = x * x * x;
    return 0.5f * x * (1.0f + tanhf(0.7978845608028654f * (x + 0.044715f * x3)));
}

// ---------------------------------------------------------------------------
// fp32 -> bf16 conversion, 4 elems/thread
// ---------------------------------------------------------------------------
__global__ __launch_bounds__(256) void cvt_f32_to_bf16(
    const float* __restrict__ src, u16* __restrict__ dst, int n4)
{
    int i = blockIdx.x * blockDim.x + threadIdx.x;
    if (i < n4) {
        v4f f = ((const v4f*)src)[i];
        v4s o;
        o.x = f2bf(f.x); o.y = f2bf(f.y); o.z = f2bf(f.z); o.w = f2bf(f.w);
        ((v4s*)dst)[i] = o;
    }
}

// ---------------------------------------------------------------------------
// WMMA GEMM:  C[M,N] = A[M,K] * B[N,K]^T   (torch Linear layout, both K-major)
// bf16 in, fp32 accumulate. Block 128x128x32, 8 wave32 waves, each wave a
// 32x64 sub-tile = 2x4 v_wmma_f32_16x16x32_bf16 accumulators.
// LDS double-buffered; fills via GLOBAL_LOAD_ASYNC_TO_LDS_B128 when available.
// Requires M%128==0, N%128==0, K%32==0 (true for every call here).
// ---------------------------------------------------------------------------
template <int STORE_BF16>
__global__ __launch_bounds__(256, 2) void gemm_bf16_nt(
    const u16* __restrict__ A, const u16* __restrict__ B,
    float* __restrict__ Cf, u16* __restrict__ Cb,
    int M, int N, int K)
{
    __shared__ u16 lA[2][128][40];   // +8 pad: kills LDS bank conflicts
    __shared__ u16 lB[2][128][40];

    const int tid  = threadIdx.x;
    const int lane = tid & 31;
    const int wid  = tid >> 5;           // 0..7
    const int wm   = wid >> 1;           // 0..3 : 32-row band
    const int wn   = wid & 1;            // 0..1 : 64-col band

    const int bn = blockIdx.x * 128;
    const int bm = blockIdx.y * 128;

    // cooperative global->LDS: each thread moves 2x16B of A and of B per tile
    const int lr = tid >> 1;             // 0..127
    const int lc = (tid & 1) * 16;       // 0 / 16

    v8f acc[2][4];
#pragma unroll
    for (int mi = 0; mi < 2; ++mi)
#pragma unroll
        for (int ni = 0; ni < 4; ++ni)
            acc[mi][ni] = (v8f){0.f,0.f,0.f,0.f,0.f,0.f,0.f,0.f};

    const u16* ga = A + (size_t)(bm + lr) * K + lc;
    const u16* gb = B + (size_t)(bn + lr) * K + lc;

    const int r16 = lane & 15;
    const int sel = lane >> 4;

    // issue one 128x32 (A and B) tile fill into buffer `buf`
    auto issue = [&](int buf, const u16* ga_, const u16* gb_) {
#if ASYNC_COPY
        __builtin_amdgcn_global_load_async_to_lds_b128(
            (GAS v4i*)ga_,       (LAS v4i*)&lA[buf][lr][lc],     0, 0);
        __builtin_amdgcn_global_load_async_to_lds_b128(
            (GAS v4i*)(ga_ + 8), (LAS v4i*)&lA[buf][lr][lc + 8], 0, 0);
        __builtin_amdgcn_global_load_async_to_lds_b128(
            (GAS v4i*)gb_,       (LAS v4i*)&lB[buf][lr][lc],     0, 0);
        __builtin_amdgcn_global_load_async_to_lds_b128(
            (GAS v4i*)(gb_ + 8), (LAS v4i*)&lB[buf][lr][lc + 8], 0, 0);
#else
        *(v4u*)&lA[buf][lr][lc]     = *(const v4u*)(ga_);
        *(v4u*)&lA[buf][lr][lc + 8] = *(const v4u*)(ga_ + 8);
        *(v4u*)&lB[buf][lr][lc]     = *(const v4u*)(gb_);
        *(v4u*)&lB[buf][lr][lc + 8] = *(const v4u*)(gb_ + 8);
#endif
    };

    int cur = 0;
    issue(0, ga, gb);                    // prologue: tile 0 in flight

    for (int k0 = 0; k0 < K; k0 += 32) {
        const u16* gan = ga + 32;
        const u16* gbn = gb + 32;
        if (k0 + 32 < K) {
            issue(cur ^ 1, gan, gbn);    // next tile in flight
            __builtin_prefetch(gan + 32, 0, 3);   // tile after next -> near cache
            __builtin_prefetch(gbn + 32, 0, 3);
#if ASYNC_COPY
            __builtin_amdgcn_s_wait_asynccnt(4);  // own 4 copies of cur done
#endif
        } else {
#if ASYNC_COPY
            __builtin_amdgcn_s_wait_asynccnt(0);
#endif
        }
        __syncthreads();                 // publish cur tile across all 8 waves
        ga = gan; gb = gbn;

        // A fragment (16x32 bf16): lanes 0-15 row M=lane, K 0..7 & 16..23;
        // lanes 16-31 same rows, K 8..15 & 24..31  (ISA 7.12.2)
        v16bf afrag[2];
#pragma unroll
        for (int mi = 0; mi < 2; ++mi) {
            const u16* p = &lA[cur][wm * 32 + mi * 16 + r16][sel * 8];
            union { v16bf v; v4u q[2]; } u;
            u.q[0] = *(const v4u*)(p);
            u.q[1] = *(const v4u*)(p + 16);
            afrag[mi] = u.v;
        }
        // B fragment (32x16 bf16): lane -> column N=lane%16, K (lane/16)*8..+7
        // & 16+(lane/16)*8..+7 (same as A) ; B row n is W[n][k] (K-contiguous)
        v16bf bfrag[4];
#pragma unroll
        for (int ni = 0; ni < 4; ++ni) {
            const u16* p = &lB[cur][wn * 64 + ni * 16 + r16][sel * 8];
            union { v16bf v; v4u q[2]; } u;
            u.q[0] = *(const v4u*)(p);
            u.q[1] = *(const v4u*)(p + 16);
            bfrag[ni] = u.v;
        }
#pragma unroll
        for (int mi = 0; mi < 2; ++mi)
#pragma unroll
            for (int ni = 0; ni < 4; ++ni)
                acc[mi][ni] = __builtin_amdgcn_wmma_f32_16x16x32_bf16(
                    false, afrag[mi], false, bfrag[ni],
                    (short)0, acc[mi][ni], false, false);
        __syncthreads();                 // all waves done reading cur
        cur ^= 1;
    }

    // epilogue: C layout (ISA): lane<16 -> M=e,N=lane ; lane>=16 -> M=e+8,N=lane-16
    const int rbase = bm + wm * 32 + (lane >> 4) * 8;
    const int cbase = bn + wn * 64 + (lane & 15);
#pragma unroll
    for (int mi = 0; mi < 2; ++mi)
#pragma unroll
        for (int ni = 0; ni < 4; ++ni)
#pragma unroll
            for (int e = 0; e < 8; ++e) {
                const int r = rbase + mi * 16 + e;
                const int c = cbase + ni * 16;
                const float v = acc[mi][ni][e];
                if (STORE_BF16) Cb[(size_t)r * N + c] = f2bf(v);
                else            Cf[(size_t)r * N + c] = v;
            }
}

// ---------------------------------------------------------------------------
// Per (b,s,h) row: top-8(left 256), top-8(right 256), top-8 of 64 cross sums,
// softmax, final index = left_i*TOPK + right_i. One wave32 per row.
// ---------------------------------------------------------------------------
__device__ __forceinline__ void select8_row(float v[8], float* outv, int* outi, int lane)
{
#pragma unroll
    for (int t = 0; t < 8; ++t) {
        float bv = v[0]; int bi = lane;
#pragma unroll
        for (int j = 1; j < 8; ++j) {
            int idx = j * 32 + lane;
            if (v[j] > bv) { bv = v[j]; bi = idx; }
        }
#pragma unroll
        for (int off = 16; off; off >>= 1) {
            float ov = __shfl_xor(bv, off, 32);
            int   oi = __shfl_xor(bi, off, 32);
            if (ov > bv || (ov == bv && oi < bi)) { bv = ov; bi = oi; }
        }
        if (lane == 0) { outv[t] = bv; outi[t] = bi; }
        if ((bi & 31) == lane) {
            const int slot = bi >> 5;
#pragma unroll
            for (int j = 0; j < 8; ++j)
                if (j == slot) v[j] = -3.402823466e38f;
        }
    }
}

__global__ __launch_bounds__(128) void topk_softmax(
    const float* __restrict__ Lg, const float* __restrict__ Rg,
    float* __restrict__ out_sc, int* __restrict__ out_ix)
{
    __shared__ float sLv[4][8]; __shared__ int sLi[4][8];
    __shared__ float sRv[4][8]; __shared__ int sRi[4][8];

    const int lane = threadIdx.x & 31;
    const int w    = threadIdx.x >> 5;
    const int row  = blockIdx.x * 4 + w;           // 0..16383

    float lv[8], rv[8];
    const float* pL = Lg + (size_t)row * NQU;
    const float* pR = Rg + (size_t)row * NQU;
#pragma unroll
    for (int j = 0; j < 8; ++j) { lv[j] = pL[j * 32 + lane]; rv[j] = pR[j * 32 + lane]; }

    select8_row(lv, sLv[w], sLi[w], lane);
    select8_row(rv, sRv[w], sRi[w], lane);

    // cross: 64 sums, each lane owns combos lane and lane+32
    float cv[2]; int cc[2];
#pragma unroll
    for (int p = 0; p < 2; ++p) {
        const int c = p * 32 + lane;
        cv[p] = sLv[w][c >> 3] + sRv[w][c & 7];
        cc[p] = c;
    }

    float wv[8]; int wc[8];
#pragma unroll
    for (int t = 0; t < 8; ++t) {
        float bv = cv[0]; int bi = cc[0];
        if (cv[1] > bv || (cv[1] == bv && cc[1] < bi)) { bv = cv[1]; bi = cc[1]; }
#pragma unroll
        for (int off = 16; off; off >>= 1) {
            float ov = __shfl_xor(bv, off, 32);
            int   oi = __shfl_xor(bi, off, 32);
            if (ov > bv || (ov == bv && oi < bi)) { bv = ov; bi = oi; }
        }
        wv[t] = bv; wc[t] = bi;
        if ((bi & 31) == lane) {
            if ((bi >> 5) == 0) cv[0] = -3.402823466e38f;
            else                cv[1] = -3.402823466e38f;
        }
    }

    // softmax (wv[0] is the max) + final indices
    float ex[8], s = 0.f;
#pragma unroll
    for (int t = 0; t < 8; ++t) { ex[t] = __expf(wv[t] - wv[0]); s += ex[t]; }
    const float inv = 1.0f / s;
    if (lane == 0) {
#pragma unroll
        for (int t = 0; t < 8; ++t) {
            const int ls = wc[t] >> 3, rs = wc[t] & 7;
            out_sc[(size_t)row * 8 + t] = ex[t] * inv;
            out_ix[(size_t)row * 8 + t] = sLi[w][ls] * TOPK + sRi[w][rs];
        }
    }
}

// ---------------------------------------------------------------------------
// Finalize: per (b,s) row — 32 expert dots vs inp_proj, GELU*score, then
// out[d] (already = residual) += sum_e w_e * emb_out[idx_e][d]
// ---------------------------------------------------------------------------
__global__ __launch_bounds__(256) void finalize_kernel(
    const float* __restrict__ proj,    // [ROWS][NHEAD*IN_F]
    const float* __restrict__ emb_in,  // [NUM_UNITS][IN_F]
    const float* __restrict__ emb_out, // [NUM_UNITS][OUT_F]
    const float* __restrict__ sc,      // [HROWS][8]
    const int*   __restrict__ ix,      // [HROWS][8]
    float* __restrict__ out)           // [ROWS][OUT_F] (residual in, result out)
{
    __shared__ float swt[32];
    __shared__ float ssc[32];
    __shared__ int   sid[32];

    const int row  = blockIdx.x;       // 0..4095
    const int tid  = threadIdx.x;
    const int lane = tid & 31;
    const int w    = tid >> 5;

    if (tid < 32) {
        sid[tid] = ix[(size_t)row * 32 + tid];
        ssc[tid] = sc[(size_t)row * 32 + tid];
    }
    __syncthreads();

    for (int e = w; e < 32; e += 8) {
        const int h = e >> 3;
        const float* pe = emb_in + (size_t)sid[e] * IN_F;
        const float* pp = proj + ((size_t)row * NHEAD + h) * IN_F;
        float d = 0.f;
        for (int i = lane; i < IN_F; i += 32) d = fmaf(pe[i], pp[i], d);
#pragma unroll
        for (int off = 16; off; off >>= 1) d += __shfl_xor(d, off, 32);
        if (lane == 0) swt[e] = ssc[e] * gelu_tanh(d);
    }
    __syncthreads();

    const int d = tid * 4;
    v4f acc = *(const v4f*)(out + (size_t)row * OUT_F + d);
#pragma unroll 4
    for (int e = 0; e < 32; ++e) {
        const float g = swt[e];
        const v4f eo = *(const v4f*)(emb_out + (size_t)sid[e] * OUT_F + d);
        acc.x = fmaf(g, eo.x, acc.x);
        acc.y = fmaf(g, eo.y, acc.y);
        acc.z = fmaf(g, eo.z, acc.z);
        acc.w = fmaf(g, eo.w, acc.w);
    }
    *(v4f*)(out + (size_t)row * OUT_F + d) = acc;
}

// ---------------------------------------------------------------------------
// Workspace layout (bytes)
// ---------------------------------------------------------------------------
constexpr size_t OFF_XBF  = 0;                                   // inp bf16  [4096][1024]
constexpr size_t OFF_WQ   = OFF_XBF  + (size_t)ROWS * IN_F * 2;  // w_query bf16 [2048][1024]
constexpr size_t OFF_WK   = OFF_WQ   + (size_t)2048 * 1024 * 2;  // w_key bf16 [4096][1024]
constexpr size_t OFF_WR   = OFF_WK   + (size_t)4096 * 1024 * 2;  // w_residual bf16 [1024][1024]
constexpr size_t OFF_WL   = OFF_WR   + (size_t)1024 * 1024 * 2;  // w_qleft bf16 [256][512]
constexpr size_t OFF_WRT  = OFF_WL   + (size_t)NQU * QDIM * 2;   // w_qright bf16 [256][512]
constexpr size_t OFF_X2   = OFF_WRT  + (size_t)NQU * QDIM * 2;   // queries bf16 [4096][2048]
constexpr size_t OFF_PROJ = OFF_X2   + (size_t)ROWS * 2048 * 2;  // inp_proj f32 [4096][4096]
constexpr size_t OFF_LG   = OFF_PROJ + (size_t)ROWS * 4096 * 4;  // left logits f32 [16384][256]
constexpr size_t OFF_RG   = OFF_LG   + (size_t)HROWS * NQU * 4;  // right logits
constexpr size_t OFF_SC   = OFF_RG   + (size_t)HROWS * NQU * 4;  // scores [16384][8]
constexpr size_t OFF_IX   = OFF_SC   + (size_t)HROWS * 8 * 4;    // indices [16384][8]

extern "C" void kernel_launch(void* const* d_in, const int* in_sizes, int n_in,
                              void* d_out, int out_size, void* d_ws, size_t ws_size,
                              hipStream_t stream)
{
    const float* inp    = (const float*)d_in[0];
    const float* w_res  = (const float*)d_in[1];
    const float* w_qry  = (const float*)d_in[2];
    const float* w_key  = (const float*)d_in[3];
    const float* w_ql   = (const float*)d_in[4];
    const float* w_qr   = (const float*)d_in[5];
    const float* embi   = (const float*)d_in[6];
    const float* embo   = (const float*)d_in[7];
    float* out = (float*)d_out;

    char* ws = (char*)d_ws;
    u16*   Xbf  = (u16*)(ws + OFF_XBF);
    u16*   Wq   = (u16*)(ws + OFF_WQ);
    u16*   Wk   = (u16*)(ws + OFF_WK);
    u16*   Wr   = (u16*)(ws + OFF_WR);
    u16*   Wl   = (u16*)(ws + OFF_WL);
    u16*   Wrt  = (u16*)(ws + OFF_WRT);
    u16*   X2   = (u16*)(ws + OFF_X2);
    float* proj = (float*)(ws + OFF_PROJ);
    float* Lg   = (float*)(ws + OFF_LG);
    float* Rg   = (float*)(ws + OFF_RG);
    float* Sc   = (float*)(ws + OFF_SC);
    int*   Ix   = (int*)(ws + OFF_IX);

    auto cvt = [&](const float* s, u16* d, size_t n) {
        int n4 = (int)(n / 4);
        cvt_f32_to_bf16<<<dim3((n4 + 255) / 256), dim3(256), 0, stream>>>(s, d, n4);
    };
    cvt(inp,   Xbf, (size_t)ROWS * IN_F);
    cvt(w_qry, Wq,  (size_t)2048 * 1024);
    cvt(w_key, Wk,  (size_t)4096 * 1024);
    cvt(w_res, Wr,  (size_t)1024 * 1024);
    cvt(w_ql,  Wl,  (size_t)NQU * QDIM);
    cvt(w_qr,  Wrt, (size_t)NQU * QDIM);

    // residual -> d_out (f32)
    gemm_bf16_nt<0><<<dim3(1024 / 128, ROWS / 128), dim3(256), 0, stream>>>(
        Xbf, Wr, out, (u16*)nullptr, ROWS, 1024, IN_F);
    // inp_proj (keys) -> proj (f32)
    gemm_bf16_nt<0><<<dim3(4096 / 128, ROWS / 128), dim3(256), 0, stream>>>(
        Xbf, Wk, proj, (u16*)nullptr, ROWS, 4096, IN_F);
    // queries -> X2 (bf16, feeds the codebook GEMMs)
    gemm_bf16_nt<1><<<dim3(2048 / 128, ROWS / 128), dim3(256), 0, stream>>>(
        Xbf, Wq, (float*)nullptr, X2, ROWS, 2048, IN_F);
    // left / right codebook logits  (X2 viewed as [16384][512])
    gemm_bf16_nt<0><<<dim3(NQU / 128, HROWS / 128), dim3(256), 0, stream>>>(
        X2, Wl, Lg, (u16*)nullptr, HROWS, NQU, QDIM);
    gemm_bf16_nt<0><<<dim3(NQU / 128, HROWS / 128), dim3(256), 0, stream>>>(
        X2, Wrt, Rg, (u16*)nullptr, HROWS, NQU, QDIM);

    topk_softmax<<<dim3(HROWS / 4), dim3(128), 0, stream>>>(Lg, Rg, Sc, Ix);

    finalize_kernel<<<dim3(ROWS), dim3(256), 0, stream>>>(proj, embi, embo, Sc, Ix, out);
}
